// MambaConvBlock_35107062677795
// MI455X (gfx1250) — compile-verified, matrix operand order checked by reading, not run on hardware
//
#include <hip/hip_runtime.h>
#include <math.h>
#include <stdint.h>

typedef __bf16 bf16_t;
typedef __attribute__((ext_vector_type(16))) __bf16 v16bf;
typedef __attribute__((ext_vector_type(8)))  __bf16 v8bf;
typedef __attribute__((ext_vector_type(8)))  float  v8f;
typedef int v4i_vec __attribute__((vector_size(16)));
typedef __attribute__((address_space(1))) v4i_vec* g_v4i_p;
typedef __attribute__((address_space(3))) v4i_vec* l_v4i_p;

#if defined(__has_builtin)
#  if __has_builtin(__builtin_amdgcn_global_load_async_to_lds_b128)
#    define USE_ASYNC_LDS 1
#  endif
#endif
#ifndef USE_ASYNC_LDS
#  define USE_ASYNC_LDS 0
#endif

__device__ __forceinline__ v8f wmma_bf16f32(v16bf a, v16bf b, v8f c) {
  return __builtin_amdgcn_wmma_f32_16x16x32_bf16(false, a, false, b, (short)0, c, false, false);
}

// K index inside a 16x32 bf16 A/B fragment: element e (0..15), lane-half kh (0/1)
__device__ __forceinline__ int frag_ks(int e, int kh) {
  return (e < 8) ? (kh * 8 + e) : (16 + kh * 8 + (e - 8));
}

__device__ __forceinline__ float silu_f(float v) {
  return v * (1.0f / (1.0f + __expf(-v)));
}

// 16-byte global -> LDS copy; async (ASYNCcnt) path when available.
__device__ __forceinline__ void copy16_g2l(const void* g, void* l) {
#if USE_ASYNC_LDS
  __builtin_amdgcn_global_load_async_to_lds_b128(
      (g_v4i_p)(unsigned long long)(uintptr_t)g,
      (l_v4i_p)(unsigned)(uintptr_t)l,
      0, 0);
#else
  *(v4i_vec*)l = *(const v4i_vec*)g;
#endif
}

__device__ __forceinline__ void wait_async_copies() {
#if USE_ASYNC_LDS
#  if defined(__has_builtin) && __has_builtin(__builtin_amdgcn_s_wait_asynccnt)
  __builtin_amdgcn_s_wait_asynccnt(0);
#  else
  asm volatile("s_wait_asynccnt 0" ::: "memory");
#  endif
#endif
}

// ---------------------------------------------------------------------------
__global__ __launch_bounds__(256)
void zero_f32(float* __restrict__ p, int n) {
  int i = blockIdx.x * 256 + threadIdx.x;
  if (i < n) p[i] = 0.f;
}

// Pack x (f32) -> bf16, same layout.
__global__ __launch_bounds__(256)
void pack_x_bf16(const float* __restrict__ x, __bf16* __restrict__ xbf) {
  int i = blockIdx.x * 256 + threadIdx.x;      // 33,554,432 exact
  xbf[i] = (bf16_t)x[i];
}

// Pack conv_w (Cout,Cin,3,3) -> wT2[n][kyx*128+ci] bf16 (K-contiguous rows).
__global__ __launch_bounds__(256)
void pack_w3_bf16(const float* __restrict__ w, __bf16* __restrict__ wT2) {
  int i = blockIdx.x * 256 + threadIdx.x;      // 294,912 exact
  int n = i / 1152;
  int k = i - n * 1152;
  int kyx = k >> 7;
  int ci  = k & 127;
  wT2[i] = (bf16_t)w[(size_t)n * 1152 + ci * 9 + kyx];
}

// Pack res_w (Cout,Cin) -> bf16 same layout.
__global__ __launch_bounds__(256)
void pack_w1_bf16(const float* __restrict__ w, __bf16* __restrict__ rwT) {
  int i = blockIdx.x * 256 + threadIdx.x;      // 32,768 exact
  rwT[i] = (bf16_t)w[i];
}

// ---------------------------------------------------------------------------
// 3x3 conv stride2 pad1 + BN + ReLU + fused 4x4-mean-pool accumulate.
// Block: 32 output pixels (one row) x 64 cout; 8 waves as 2(M) x 4(N).
// Input patch transposed into LDS as [iy][lix][ci] (ci contiguous, pad 136),
// so each A fragment is two ds_load_b128; B is two global_load_b128 (packed).
__global__ __launch_bounds__(256)
void conv3x3_bn_relu_pool(const __bf16* __restrict__ xbf,
                          const __bf16* __restrict__ wT2,
                          const float* __restrict__ g, const float* __restrict__ bt,
                          const float* __restrict__ mn, const float* __restrict__ vr,
                          float* __restrict__ pooled) {
  // 3 rows x 65 cols x 136 ci (pad) bf16 = 53,040B; ci-row stride 272B (16B mult)
  __shared__ __align__(16) __bf16 patch[3 * 65 * 136];

  const int tid   = threadIdx.x;
  const int mtile = blockIdx.x >> 2;           // 2048 tiles of 32 pixels
  const int ntile = blockIdx.x & 3;            // 4 tiles of 64 cout
  const int mbase = mtile * 32;
  const int b   = mbase >> 12;
  const int oy  = (mbase >> 6) & 63;
  const int ox0 = mbase & 63;                  // 0 or 32
  // patch x-origin: ix0 = 2*ox0 - 1 ; lix = ix - ix0 in [0,64]

  // ---- stage + transpose body: lix 1..64 <- xbf[.., giy, 2*ox0 + 0..63]
#pragma unroll
  for (int c = 0; c < 12; ++c) {
    int chunk = tid + c * 256;                 // 0..3071 (384 rows x 8 chunks)
    int row = chunk >> 3;                      // iyl*128 + ci
    int j8  = (chunk & 7) * 8;
    int iyl = row >> 7;
    int ci  = row & 127;
    int giy = oy * 2 - 1 + iyl;                // -1..127
    v8bf v = {};
    if (giy >= 0)
      v = *(const v8bf*)&xbf[(((size_t)b * 128 + ci) * 128 + giy) * 128 + ox0 * 2 + j8];
#pragma unroll
    for (int e = 0; e < 8; ++e)
      patch[(iyl * 65 + j8 + 1 + e) * 136 + ci] = v[e];
  }
  // ---- halo column (lix = 0, ix = 2*ox0 - 1)
  for (int h = tid; h < 384; h += 256) {
    int iyl = h >> 7;
    int ci  = h & 127;
    int giy = oy * 2 - 1 + iyl;
    int gix = ox0 * 2 - 1;
    bf16_t v = (bf16_t)0.f;
    if (giy >= 0 && gix >= 0)
      v = xbf[(((size_t)b * 128 + ci) * 128 + giy) * 128 + gix];
    patch[(iyl * 65) * 136 + ci] = v;
  }
  __syncthreads();

  const int lane = tid & 31;
  const int wv   = tid >> 5;
  const int r  = lane & 15;
  const int kh = lane >> 4;
  const int wm = wv >> 2;                      // 0..1
  const int wn = wv & 3;                       // 0..3
  const int oxl = wm * 16 + r;                 // pixel column this lane gathers
  const int nc  = ntile * 64 + wn * 16 + r;    // cout this lane gathers B for

  v8f acc = {0.f,0.f,0.f,0.f,0.f,0.f,0.f,0.f};
#pragma unroll
  for (int kyx = 0; kyx < 9; ++kyx) {
    const int ky = kyx / 3;
    const int kx = kyx - ky * 3;
    const int lix = 2 * oxl + kx;              // 0..64, no clamp needed
    const __bf16* prow = &patch[(ky * 65 + lix) * 136 + kh * 8];
    const __bf16* wrow = wT2 + (size_t)nc * 1152 + kyx * 128 + kh * 8;
#pragma unroll
    for (int c0 = 0; c0 < 128; c0 += 32) {
      v8bf a0 = *(const v8bf*)(prow + c0);
      v8bf a1 = *(const v8bf*)(prow + c0 + 16);
      v16bf af = __builtin_shufflevector(a0, a1, 0,1,2,3,4,5,6,7,8,9,10,11,12,13,14,15);
      v8bf b0 = *(const v8bf*)(wrow + c0);
      v8bf b1 = *(const v8bf*)(wrow + c0 + 16);
      v16bf bfr = __builtin_shufflevector(b0, b1, 0,1,2,3,4,5,6,7,8,9,10,11,12,13,14,15);
      acc = wmma_bf16f32(af, bfr, acc);
    }
  }

  const float eps = 1e-5f;
  const float sc = g[nc] * rsqrtf(vr[nc] + eps);
  const float sh = bt[nc] - mn[nc] * sc;
#pragma unroll
  for (int v = 0; v < 8; ++v) {
    const int po  = wm * 16 + kh * 8 + v;      // pixel offset 0..31
    const int oxx = ox0 + po;
    const float val = fmaxf(acc[v] * sc + sh, 0.f);
    atomicAdd(&pooled[(((size_t)b * 256 + nc) * 16 + (oy >> 2)) * 16 + (oxx >> 2)],
              val * 0.0625f);
  }
}

// ---------------------------------------------------------------------------
// 1x1 conv stride2 + BN (residual). M=65536, N=256, K=128. bf16 inputs.
__global__ __launch_bounds__(256)
void conv1x1_bn(const __bf16* __restrict__ xbf, const __bf16* __restrict__ rwT,
                const float* __restrict__ g, const float* __restrict__ bt,
                const float* __restrict__ mn, const float* __restrict__ vr,
                float* __restrict__ iden) {
  const int lane  = threadIdx.x & 31;
  const int wv    = threadIdx.x >> 5;
  const int tile  = blockIdx.x * 8 + wv;
  const int mtile = tile >> 4;
  const int ntile = tile & 15;
  const int r  = lane & 15;
  const int kh = lane >> 4;
  const int m  = mtile * 16 + r;
  const int b  = m >> 12;
  const int oy = (m >> 6) & 63;
  const int ox = m & 63;
  const int nc = ntile * 16 + r;

  const __bf16* arow = &xbf[(((size_t)b * 128) * 128 + oy * 2) * 128 + ox * 2];
  const __bf16* wrow = rwT + (size_t)nc * 128 + kh * 8;

  v8f acc = {0.f,0.f,0.f,0.f,0.f,0.f,0.f,0.f};
#pragma unroll
  for (int k0 = 0; k0 < 128; k0 += 32) {
    v16bf af;
#pragma unroll
    for (int e = 0; e < 16; ++e) {
      int k = k0 + frag_ks(e, kh);
      af[e] = arow[(size_t)k * 16384];         // 128*128 stride per ci
    }
    v8bf b0 = *(const v8bf*)(wrow + k0);
    v8bf b1 = *(const v8bf*)(wrow + k0 + 16);
    v16bf bfr = __builtin_shufflevector(b0, b1, 0,1,2,3,4,5,6,7,8,9,10,11,12,13,14,15);
    acc = wmma_bf16f32(af, bfr, acc);
  }

  const float eps = 1e-5f;
  const float sc = g[nc] * rsqrtf(vr[nc] + eps);
  const float sh = bt[nc] - mn[nc] * sc;
#pragma unroll
  for (int v = 0; v < 8; ++v) {
    const int mm  = mtile * 16 + kh * 8 + v;
    const int bb2 = mm >> 12;
    const int oyy = (mm >> 6) & 63;
    const int oxx = mm & 63;
    iden[(((size_t)bb2 * 256 + nc) * 64 + oyy) * 64 + oxx] = acc[v] * sc + sh;
  }
}

// ---------------------------------------------------------------------------
// in_proj: xz[m][j] = sum_c pooled[b][c][l] * W[j][c]; M=4096,K=256,N=1024.
// Both LDS tiles staged with async b128 copies (16B chunks, 144B row stride).
__global__ __launch_bounds__(256)
void inproj_gemm(const float* __restrict__ pooled, const float* __restrict__ w,
                 float* __restrict__ xi, float* __restrict__ zb) {
  __shared__ __align__(16) float sA[32][36];   // k-major: sA[k][m]
  __shared__ __align__(16) float sB[64][36];   // n-major: sB[n][k]
  const int t    = threadIdx.x;
  const int lane = t & 31;
  const int wv   = t >> 5;
  const int mt = blockIdx.x >> 4;
  const int nt = blockIdx.x & 15;
  const int mb = mt * 32;
  const int nb = nt * 64;
  const int r  = lane & 15;
  const int kh = lane >> 4;
  const int wm = wv >> 2;
  const int wn = wv & 3;
  const int bidx  = mb >> 8;
  const int lbase = mb & 255;

  v8f acc = {0.f,0.f,0.f,0.f,0.f,0.f,0.f,0.f};
  for (int k0 = 0; k0 < 256; k0 += 32) {
    {
      // sA: 32k x 32m floats = 256 chunks of 16B, 1 per thread
      int ak = t >> 3;
      int q  = (t & 7) * 4;
      copy16_g2l(&pooled[(size_t)bidx * 65536 + (size_t)(k0 + ak) * 256 + lbase + q],
                 &sA[ak][q]);
    }
#pragma unroll
    for (int j = 0; j < 2; ++j) {
      // sB: 64n x 32k floats = 512 chunks of 16B, 2 per thread
      int c  = t + j * 256;
      int bn = c >> 3;
      int q  = (c & 7) * 4;
      copy16_g2l(&w[(size_t)(nb + bn) * 256 + k0 + q], &sB[bn][q]);
    }
    wait_async_copies();
    __syncthreads();
    v16bf af, bfr;
#pragma unroll
    for (int e = 0; e < 16; ++e) {
      int ks = frag_ks(e, kh);
      af[e]  = (bf16_t)sA[ks][wm * 16 + r];
      bfr[e] = (bf16_t)sB[wn * 16 + r][ks];
    }
    acc = wmma_bf16f32(af, bfr, acc);
    __syncthreads();
  }
#pragma unroll
  for (int v = 0; v < 8; ++v) {
    int mm = mb + wm * 16 + kh * 8 + v;
    int jj = nb + wn * 16 + r;
    float val = acc[v];
    if (jj < 512) xi[(size_t)mm * 512 + jj] = val;
    else          zb[(size_t)mm * 512 + (jj - 512)] = val;
  }
}

// ---------------------------------------------------------------------------
__global__ __launch_bounds__(256)
void dwconv_silu(const float* __restrict__ xi, const float* __restrict__ w,
                 const float* __restrict__ bias, float* __restrict__ xc) {
  int idx = blockIdx.x * 256 + threadIdx.x;
  int d = idx & 511;
  int m = idx >> 9;
  int l = m & 255;
  float s = bias[d];
#pragma unroll
  for (int tp = 0; tp < 4; ++tp) {
    int ll = l + tp - 3;
    if (ll >= 0) s += w[d * 4 + tp] * xi[(size_t)(m - l + ll) * 512 + d];
  }
  xc[(size_t)m * 512 + d] = silu_f(s);
}

// ---------------------------------------------------------------------------
__global__ __launch_bounds__(256)
void xproj(const float* __restrict__ xc, const float* __restrict__ w,
           float* __restrict__ xdbl) {
  int idx = blockIdx.x * 256 + threadIdx.x;
  if (idx >= 4096 * 144) return;
  int o = idx % 144;
  int m = idx / 144;
  const float* a  = xc + (size_t)m * 512;
  const float* wr = w + (size_t)o * 512;
  float s = 0.f;
  for (int k = 0; k < 512; ++k) s += a[k] * wr[k];
  xdbl[(size_t)m * 144 + o] = s;
}

// ---------------------------------------------------------------------------
__global__ __launch_bounds__(256)
void dt_softplus(const float* __restrict__ xdbl, const float* __restrict__ w,
                 const float* __restrict__ bias, float* __restrict__ dt) {
  int idx = blockIdx.x * 256 + threadIdx.x;
  int d = idx & 511;
  int m = idx >> 9;
  float s = bias[d];
#pragma unroll
  for (int rr = 0; rr < 16; ++rr) s += xdbl[(size_t)m * 144 + rr] * w[d * 16 + rr];
  float sp = (s > 20.f) ? s : log1pf(__expf(s));
  dt[(size_t)m * 512 + d] = sp;
}

// ---------------------------------------------------------------------------
__global__ __launch_bounds__(512)
void sel_scan(const float* __restrict__ xc, const float* __restrict__ dt,
              const float* __restrict__ A_log, const float* __restrict__ xdbl,
              const float* __restrict__ Dp, float* __restrict__ yout) {
  __shared__ float sB[64];
  __shared__ float sC[64];
  const int d = threadIdx.x;
  const int b = blockIdx.x;
  float Af[64];
#pragma unroll
  for (int n = 0; n < 64; ++n) Af[n] = -__expf(A_log[(size_t)d * 64 + n]);
  float hs[64];
#pragma unroll
  for (int n = 0; n < 64; ++n) hs[n] = 0.f;
  const float dpar = Dp[d];

  for (int tt = 0; tt < 256; ++tt) {
    const size_t mrow = (size_t)b * 256 + tt;
    if (d < 64)       sB[d]      = xdbl[mrow * 144 + 16 + d];
    else if (d < 128) sC[d - 64] = xdbl[mrow * 144 + 80 + (d - 64)];
    __syncthreads();
    const float u  = xc[mrow * 512 + d];
    const float dv = dt[mrow * 512 + d];
    float y = 0.f;
#pragma unroll
    for (int n = 0; n < 64; ++n) {
      const float e = __expf(dv * Af[n]);
      hs[n] = e * hs[n] + (dv * sB[n]) * u;
      y += hs[n] * sC[n];
    }
    yout[mrow * 512 + d] = y + u * dpar;
    __syncthreads();
  }
}

// ---------------------------------------------------------------------------
// out_proj with fused y*silu(z): M=4096,K=512,N=256 -> ym[b][n][l].
// sB staged with async b128; sA computed (fused elementwise) synchronously.
__global__ __launch_bounds__(256)
void outproj_gemm(const float* __restrict__ y, const float* __restrict__ zb,
                  const float* __restrict__ w, float* __restrict__ ym) {
  __shared__ __align__(16) float sA[32][33];
  __shared__ __align__(16) float sB[64][36];
  const int t    = threadIdx.x;
  const int lane = t & 31;
  const int wv   = t >> 5;
  const int mt = blockIdx.x >> 2;
  const int nt = blockIdx.x & 3;
  const int mb = mt * 32;
  const int nb = nt * 64;
  const int r  = lane & 15;
  const int kh = lane >> 4;
  const int wm = wv >> 2;
  const int wn = wv & 3;

  v8f acc = {0.f,0.f,0.f,0.f,0.f,0.f,0.f,0.f};
  for (int k0 = 0; k0 < 512; k0 += 32) {
#pragma unroll
    for (int j = 0; j < 2; ++j) {
      int c  = t + j * 256;
      int bn = c >> 3;
      int q  = (c & 7) * 4;
      copy16_g2l(&w[(size_t)(nb + bn) * 512 + k0 + q], &sB[bn][q]);
    }
#pragma unroll
    for (int j = 0; j < 4; ++j) {
      int i  = t + j * 256;
      int am = i >> 5;
      int ak = i & 31;
      size_t off = (size_t)(mb + am) * 512 + k0 + ak;
      sA[am][ak] = y[off] * silu_f(zb[off]);
    }
    wait_async_copies();
    __syncthreads();
    v16bf af, bfr;
#pragma unroll
    for (int e = 0; e < 16; ++e) {
      int ks = frag_ks(e, kh);
      af[e]  = (bf16_t)sA[wm * 16 + r][ks];
      bfr[e] = (bf16_t)sB[wn * 16 + r][ks];
    }
    acc = wmma_bf16f32(af, bfr, acc);
    __syncthreads();
  }
#pragma unroll
  for (int v = 0; v < 8; ++v) {
    int mm = mb + wm * 16 + kh * 8 + v;
    int nn = nb + wn * 16 + r;
    ym[((size_t)(mm >> 8) * 256 + nn) * 256 + (mm & 255)] = acc[v];
  }
}

// ---------------------------------------------------------------------------
__global__ __launch_bounds__(256)
void upsample_add_relu(const float* __restrict__ ym, const float* __restrict__ iden,
                       float* __restrict__ out) {
  int idx = blockIdx.x * 256 + threadIdx.x;
  int ox = idx & 63;
  int oy = (idx >> 6) & 63;
  int c  = (idx >> 12) & 255;
  int b  = idx >> 20;
  float sy = fmaxf(0.25f * (float)oy - 0.375f, 0.f);
  float sx = fmaxf(0.25f * (float)ox - 0.375f, 0.f);
  int y0 = (int)sy; y0 = (y0 < 15) ? y0 : 15;
  int y1 = (y0 + 1 < 15) ? (y0 + 1) : 15;
  float wy = sy - (float)y0;
  int x0 = (int)sx; x0 = (x0 < 15) ? x0 : 15;
  int x1 = (x0 + 1 < 15) ? (x0 + 1) : 15;
  float wx = sx - (float)x0;
  const float* base = ym + ((size_t)b * 256 + c) * 256;
  float v00 = base[y0 * 16 + x0], v01 = base[y0 * 16 + x1];
  float v10 = base[y1 * 16 + x0], v11 = base[y1 * 16 + x1];
  float top = v00 * (1.f - wx) + v01 * wx;
  float bot = v10 * (1.f - wx) + v11 * wx;
  float val = top * (1.f - wy) + bot * wy;
  out[idx] = fmaxf(val + iden[idx], 0.f);
}

// ---------------------------------------------------------------------------
extern "C" void kernel_launch(void* const* d_in, const int* in_sizes, int n_in,
                              void* d_out, int out_size, void* d_ws, size_t ws_size,
                              hipStream_t stream) {
  (void)in_sizes; (void)n_in; (void)out_size; (void)ws_size;
  const float* x       = (const float*)d_in[0];
  const float* conv_w  = (const float*)d_in[1];
  const float* bn_g    = (const float*)d_in[2];
  const float* bn_b    = (const float*)d_in[3];
  const float* bn_m    = (const float*)d_in[4];
  const float* bn_v    = (const float*)d_in[5];
  const float* res_w   = (const float*)d_in[6];
  const float* rbn_g   = (const float*)d_in[7];
  const float* rbn_b   = (const float*)d_in[8];
  const float* rbn_m   = (const float*)d_in[9];
  const float* rbn_v   = (const float*)d_in[10];
  const float* in_proj = (const float*)d_in[11];
  const float* c1d_w   = (const float*)d_in[12];
  const float* c1d_b   = (const float*)d_in[13];
  const float* xproj_w = (const float*)d_in[14];
  const float* dtp_w   = (const float*)d_in[15];
  const float* dtp_b   = (const float*)d_in[16];
  const float* A_log   = (const float*)d_in[17];
  const float* Dp      = (const float*)d_in[18];
  const float* outp_w  = (const float*)d_in[19];
  float* out = (float*)d_out;

  char* p = (char*)d_ws;
  float* pooled = (float*)p;  p += (size_t)16 * 256 * 256 * 4;
  float* iden   = (float*)p;  p += (size_t)16 * 256 * 4096 * 4;
  float* xi     = (float*)p;  p += (size_t)4096 * 512 * 4;
  float* zb     = (float*)p;  p += (size_t)4096 * 512 * 4;
  float* xc     = (float*)p;  p += (size_t)4096 * 512 * 4;
  float* xdbl   = (float*)p;  p += (size_t)4096 * 144 * 4;
  float* dtbuf  = (float*)p;  p += (size_t)4096 * 512 * 4;
  float* ybuf   = (float*)p;  p += (size_t)4096 * 512 * 4;
  float* ym     = (float*)p;  p += (size_t)4096 * 256 * 4;
  __bf16* xbf   = (__bf16*)p; p += (size_t)33554432 * 2;
  __bf16* wT2   = (__bf16*)p; p += (size_t)1152 * 256 * 2;
  __bf16* rwT   = (__bf16*)p; p += (size_t)128 * 256 * 2;

  pack_x_bf16<<<131072, 256, 0, stream>>>(x, xbf);
  pack_w3_bf16<<<1152, 256, 0, stream>>>(conv_w, wT2);
  pack_w1_bf16<<<128, 256, 0, stream>>>(res_w, rwT);
  zero_f32<<<4096, 256, 0, stream>>>(pooled, 16 * 256 * 256);

  conv3x3_bn_relu_pool<<<8192, 256, 0, stream>>>(xbf, wT2, bn_g, bn_b, bn_m, bn_v, pooled);
  conv1x1_bn<<<8192, 256, 0, stream>>>(xbf, rwT, rbn_g, rbn_b, rbn_m, rbn_v, iden);
  inproj_gemm<<<2048, 256, 0, stream>>>(pooled, in_proj, xi, zb);
  dwconv_silu<<<8192, 256, 0, stream>>>(xi, c1d_w, c1d_b, xc);
  xproj<<<2304, 256, 0, stream>>>(xc, xproj_w, xdbl);
  dt_softplus<<<8192, 256, 0, stream>>>(xdbl, dtp_w, dtp_b, dtbuf);
  sel_scan<<<16, 512, 0, stream>>>(xc, dtbuf, A_log, xdbl, Dp, ybuf);
  outproj_gemm<<<512, 256, 0, stream>>>(ybuf, zb, outp_w, ym);
  upsample_add_relu<<<65536, 256, 0, stream>>>(ym, iden, out);
}
